// SparseHypergraphLayer_49615462203488
// MI455X (gfx1250) — matrix operand order repaired and checked by reading, not compiled, not run on hardware
//
#include <hip/hip_runtime.h>
#include <math.h>

typedef __attribute__((ext_vector_type(16))) _Float16 v16h;
typedef __attribute__((ext_vector_type(8)))  _Float16 v8h;
typedef __attribute__((ext_vector_type(8)))  float    v8f;
typedef int v4i __attribute__((vector_size(4 * sizeof(int))));
typedef __attribute__((address_space(1))) v4i gv4i;   // global
typedef __attribute__((address_space(3))) v4i lv4i;   // LDS

#define FDIM 128
#define APITCH 136   // LDS pitch in f16: row stride 272B -> bank 4*l mod 64, conflict-free

#if __has_builtin(__builtin_amdgcn_global_load_async_to_lds_b128) && \
    __has_builtin(__builtin_amdgcn_s_wait_asynccnt)
#define ASYNC_LDS 1
#else
#define ASYNC_LDS 0
#endif

__device__ __forceinline__ v16h cat8(v8h lo, v8h hi) {
  return __builtin_shufflevector(lo, hi, 0, 1, 2, 3, 4, 5, 6, 7, 8, 9, 10, 11,
                                 12, 13, 14, 15);
}

// ---------------------------------------------------------------------------
// GEMM: [R,128] x [128,128] -> [R,128], rows 0..N-1 from x, N..N+M-1 from ha.
// Block = 16 rows x 128 cols; 8 waves, one 16x16 WMMA tile each, K looped 4x32.
// W staged TRANSPOSED in LDS so B fragments are contiguous b128 reads.
// ---------------------------------------------------------------------------
__global__ __launch_bounds__(256) void k_gemm_wmma(
    const float* __restrict__ x, const float* __restrict__ ha,
    const float* __restrict__ W, float* __restrict__ out, int N, int M) {
  __shared__ __align__(16) _Float16 sWt[FDIM * APITCH];  // sWt[n*APITCH + k]
  __shared__ __align__(16) _Float16 sA[16 * APITCH];     // sA[m*APITCH + k]
  const int R = N + M;
  const int row0 = blockIdx.x * 16;
  const int t = threadIdx.x;

#if ASYNC_LDS
  // async-copy raw f32 W into LDS (ASYNCcnt path), then convert+transpose
  __shared__ __align__(16) float sWf[FDIM * FDIM];
  for (int i = t; i < (FDIM * FDIM) / 4; i += 256) {
    __builtin_amdgcn_global_load_async_to_lds_b128(
        (gv4i*)(W + i * 4), (lv4i*)(sWf + i * 4), 0, 0);
  }
  __builtin_amdgcn_s_wait_asynccnt(0);
  __syncthreads();
  for (int i = t; i < FDIM * FDIM; i += 256) {
    int k = i >> 7, n = i & 127;
    sWt[n * APITCH + k] = (_Float16)sWf[k * FDIM + n];
  }
#else
  for (int i = t; i < FDIM * FDIM; i += 256) {
    int k = i >> 7, n = i & 127;              // coalesced global read of W[k][n]
    sWt[n * APITCH + k] = (_Float16)W[i];     // transposed LDS store
  }
#endif
  // stage 16 A rows (f32 -> f16)
  for (int i = t; i < 16 * FDIM; i += 256) {
    int m = i >> 7, k = i & 127;
    int r = row0 + m;
    float v = 0.f;
    if (r < N)      v = x[(size_t)r * FDIM + k];
    else if (r < R) v = ha[(size_t)(r - N) * FDIM + k];
    sA[m * APITCH + k] = (_Float16)v;
  }
  __syncthreads();

  const int wave = t >> 5;        // column tile 0..7
  const int lane = t & 31;
  const int half = lane >> 4;     // 0/1
  const int l16  = lane & 15;
  const int coln = wave * 16 + l16;
  const _Float16* ar = &sA[l16 * APITCH];
  const _Float16* br = &sWt[coln * APITCH];

  v8f c = {};
#pragma unroll
  for (int kc = 0; kc < 4; ++kc) {
    const int k0 = kc * 32;
    // A 16x32 f16: lane<16 -> K {k0..k0+7, k0+16..k0+23}; lane>=16 -> +8
    v8h alo = *(const v8h*)(ar + k0 + half * 8);
    v8h ahi = *(const v8h*)(ar + k0 + 16 + half * 8);
    // B 32x16 f16 (W transposed in LDS): lane<16 -> K {k0..k0+7, k0+16..k0+23}; lane>=16 -> +8
    v8h blo = *(const v8h*)(br + k0 + half * 8);
    v8h bhi = *(const v8h*)(br + k0 + 16 + half * 8);
    c = __builtin_amdgcn_wmma_f32_16x16x32_f16(false, cat8(alo, ahi), false,
                                               cat8(blo, bhi), (short)0, c,
                                               false, false);
  }
  // C/D: VGPR r -> row r + 8*half, col = lane%16.
  // Block-uniform guard; fast path = 8 plain stores with immediate offsets.
  float* orow = out + (size_t)(row0 + 8 * half) * FDIM + coln;
  if (row0 + 16 <= R) {
#pragma unroll
    for (int r = 0; r < 8; ++r) orow[(size_t)r * FDIM] = c[r];
  } else {
#pragma unroll
    for (int r = 0; r < 8; ++r)
      if (row0 + r + 8 * half < R) orow[(size_t)r * FDIM] = c[r];
  }
}

// ---------------------------------------------------------------------------
// a_node / a_edge: one wave per row, 4 elems per lane + shfl reduction.
// ---------------------------------------------------------------------------
__global__ __launch_bounds__(256) void k_dot_att(
    const float* __restrict__ xw_ea, const float* __restrict__ att,
    float* __restrict__ avec, int N, int R) {
  int wave = threadIdx.x >> 5;
  int lane = threadIdx.x & 31;
  int r = blockIdx.x * 8 + wave;
  if (r >= R) return;
  const float* arow = att + ((r < N) ? 0 : FDIM);
  float s = 0.f;
#pragma unroll
  for (int j = 0; j < 4; ++j) {
    int f = lane + 32 * j;
    s += xw_ea[(size_t)r * FDIM + f] * arow[f];
  }
#pragma unroll
  for (int off = 16; off > 0; off >>= 1) s += __shfl_xor(s, off, 32);
  if (lane == 0) avec[r] = s;
}

// ---------------------------------------------------------------------------
__global__ __launch_bounds__(256) void k_deg(
    const int* __restrict__ node_idx, const int* __restrict__ edge_idx,
    float* __restrict__ Dd, float* __restrict__ Bd, int E) {
  int e = blockIdx.x * 256 + threadIdx.x;
  if (e >= E) return;
  atomicAdd(&Dd[node_idx[e]], 1.0f);
  atomicAdd(&Bd[edge_idx[e]], 1.0f);
}

// order-preserving float<->uint for atomic max
__device__ __forceinline__ unsigned enc_ord(float f) {
  unsigned u = __float_as_uint(f);
  return (u & 0x80000000u) ? ~u : (u | 0x80000000u);
}
__device__ __forceinline__ float dec_ord(unsigned u) {
  return (u & 0x80000000u) ? __uint_as_float(u & 0x7FFFFFFFu)
                           : __uint_as_float(~u);
}

__global__ __launch_bounds__(256) void k_alpha_max(
    const int* __restrict__ node_idx, const int* __restrict__ edge_idx,
    const float* __restrict__ avec, float* __restrict__ alpha,
    unsigned* __restrict__ segmax, int N, int E) {
  int e = blockIdx.x * 256 + threadIdx.x;
  if (e >= E) return;
  float al = avec[node_idx[e]] + avec[N + edge_idx[e]];
  al = (al > 0.f) ? al : 0.2f * al;
  alpha[e] = al;
  atomicMax(&segmax[edge_idx[e]], enc_ord(al));
}

__global__ __launch_bounds__(256) void k_alpha_exp(
    const int* __restrict__ edge_idx, float* __restrict__ alpha,
    const unsigned* __restrict__ segmax, float* __restrict__ segsum, int E) {
  int e = blockIdx.x * 256 + threadIdx.x;
  if (e >= E) return;
  int seg = edge_idx[e];
  float ex = expf(alpha[e] - dec_ord(segmax[seg]));
  alpha[e] = ex;
  atomicAdd(&segsum[seg], ex);
}

// node -> hyperedge scatter: 128 threads per incidence, persistent blocks,
// prefetch next iteration's gather row (global_prefetch_b8).
__global__ __launch_bounds__(256) void k_scatter1(
    const int* __restrict__ node_idx, const int* __restrict__ edge_idx,
    const float* __restrict__ alpha, const float* __restrict__ segsum,
    const float* __restrict__ Bd, const float* __restrict__ xw,
    float* __restrict__ efeat, int E) {
  const int sub = threadIdx.x >> 7;
  const int f = threadIdx.x & 127;
  const int stride = gridDim.x * 2;
  for (int e = blockIdx.x * 2 + sub; e < E; e += stride) {
    int node = node_idx[e], edge = edge_idx[e];
    int en = e + stride;
    if (en < E)
      __builtin_prefetch(&xw[(size_t)node_idx[en] * FDIM + f], 0, 1);
    float bsum = Bd[edge];
    float binv = (bsum > 0.f) ? (1.f / bsum) : 0.f;
    float wgt = binv * alpha[e] / (segsum[edge] + 1e-16f);
    atomicAdd(&efeat[(size_t)edge * FDIM + f],
              wgt * xw[(size_t)node * FDIM + f]);
  }
}

// hyperedge -> node scatter
__global__ __launch_bounds__(256) void k_scatter2(
    const int* __restrict__ node_idx, const int* __restrict__ edge_idx,
    const float* __restrict__ alpha, const float* __restrict__ segsum,
    const float* __restrict__ Dd, const float* __restrict__ efeat,
    float* __restrict__ acc, int E) {
  const int sub = threadIdx.x >> 7;
  const int f = threadIdx.x & 127;
  const int stride = gridDim.x * 2;
  for (int e = blockIdx.x * 2 + sub; e < E; e += stride) {
    int node = node_idx[e], edge = edge_idx[e];
    int en = e + stride;
    if (en < E)
      __builtin_prefetch(&efeat[(size_t)edge_idx[en] * FDIM + f], 0, 1);
    float dsum = Dd[node];
    float dinv = (dsum > 0.f) ? (1.f / dsum) : 0.f;
    float wgt = dinv * alpha[e] / (segsum[edge] + 1e-16f);
    atomicAdd(&acc[(size_t)node * FDIM + f],
              wgt * efeat[(size_t)edge * FDIM + f]);
  }
}

// BN column statistics (thread = feature, coalesced rows, grid-stride)
__global__ __launch_bounds__(128) void k_bnstats(
    const float* __restrict__ acc, const float* __restrict__ bias,
    float* __restrict__ colsum, float* __restrict__ colsumsq, int N) {
  int f = threadIdx.x;
  float b = bias[f];
  float s = 0.f, sq = 0.f;
  for (int r = blockIdx.x; r < N; r += gridDim.x) {
    float v = acc[(size_t)r * FDIM + f] + b;
    s += v;
    sq += v * v;
  }
  atomicAdd(&colsum[f], s);
  atomicAdd(&colsumsq[f], sq);
}

__global__ __launch_bounds__(128) void k_bnparams(
    const float* __restrict__ colsum, const float* __restrict__ colsumsq,
    const float* __restrict__ gamma, const float* __restrict__ beta,
    float* __restrict__ scale, float* __restrict__ shift, int N) {
  int f = threadIdx.x;
  float invN = 1.f / (float)N;
  float mu = colsum[f] * invN;
  float var = colsumsq[f] * invN - mu * mu;
  float sc = gamma[f] * rsqrtf(var + 1e-5f);
  scale[f] = sc;
  shift[f] = beta[f] - mu * sc;
}

// scale/shift + ELU + residual
__global__ __launch_bounds__(256) void k_final(
    const float* __restrict__ acc, const float* __restrict__ bias,
    const float* __restrict__ scale, const float* __restrict__ shift,
    const float* __restrict__ x, float* __restrict__ out, long long total) {
  long long i = (long long)blockIdx.x * 256 + threadIdx.x;
  if (i >= total) return;
  int f = (int)(i & 127);
  float v = acc[i] + bias[f];
  float y = scale[f] * v + shift[f];
  y = (y > 0.f) ? y : expm1f(y);
  out[i] = y + x[i];
}

// ---------------------------------------------------------------------------
extern "C" void kernel_launch(void* const* d_in, const int* in_sizes, int n_in,
                              void* d_out, int out_size, void* d_ws,
                              size_t ws_size, hipStream_t stream) {
  const float* x     = (const float*)d_in[0];
  const float* ha    = (const float*)d_in[1];
  const float* W     = (const float*)d_in[2];
  const float* att   = (const float*)d_in[3];
  const float* bias  = (const float*)d_in[4];
  const float* gamma = (const float*)d_in[5];
  const float* beta  = (const float*)d_in[6];
  const int* node_idx = (const int*)d_in[7];
  const int* edge_idx = (const int*)d_in[8];
  float* out = (float*)d_out;

  const int N = in_sizes[0] / FDIM;
  const int M = in_sizes[1] / FDIM;
  const int E = in_sizes[7];
  const int R = N + M;

  char* p = (char*)d_ws;
  float* xw_ea = (float*)p;           p += (size_t)R * FDIM * 4;
  float* avec  = (float*)p;           p += (size_t)R * 4;
  float* alpha = (float*)p;           p += (size_t)E * 4;
  unsigned* segmax = (unsigned*)p;    p += (size_t)M * 4;
  float* segsum = (float*)p;          p += (size_t)M * 4;
  float* Dd    = (float*)p;           p += (size_t)N * 4;
  float* Bd    = (float*)p;           p += (size_t)M * 4;
  float* efeat = (float*)p;           p += (size_t)M * FDIM * 4;
  float* acc   = (float*)p;           p += (size_t)N * FDIM * 4;
  float* colsum   = (float*)p;        p += 128 * 4;
  float* colsumsq = (float*)p;        p += 128 * 4;
  float* scale = (float*)p;           p += 128 * 4;
  float* shift = (float*)p;           p += 128 * 4;
  if ((size_t)(p - (char*)d_ws) > ws_size) return;

  (void)hipMemsetAsync(segmax, 0, (size_t)M * 4, stream);   // enc_ord minimum
  (void)hipMemsetAsync(segsum, 0, (size_t)M * 4, stream);
  (void)hipMemsetAsync(Dd, 0, (size_t)N * 4, stream);
  (void)hipMemsetAsync(Bd, 0, (size_t)M * 4, stream);
  (void)hipMemsetAsync(efeat, 0, (size_t)M * FDIM * 4, stream);
  (void)hipMemsetAsync(acc, 0, (size_t)N * FDIM * 4, stream);
  (void)hipMemsetAsync(colsum, 0, 256 * 4, stream);  // colsum + colsumsq

  k_gemm_wmma<<<(R + 15) / 16, 256, 0, stream>>>(x, ha, W, xw_ea, N, M);
  k_dot_att<<<(R + 7) / 8, 256, 0, stream>>>(xw_ea, att, avec, N, R);
  k_deg<<<(E + 255) / 256, 256, 0, stream>>>(node_idx, edge_idx, Dd, Bd, E);
  k_alpha_max<<<(E + 255) / 256, 256, 0, stream>>>(node_idx, edge_idx, avec,
                                                   alpha, segmax, N, E);
  k_alpha_exp<<<(E + 255) / 256, 256, 0, stream>>>(edge_idx, alpha, segmax,
                                                   segsum, E);
  k_scatter1<<<8192, 256, 0, stream>>>(node_idx, edge_idx, alpha, segsum, Bd,
                                       xw_ea, efeat, E);
  k_scatter2<<<8192, 256, 0, stream>>>(node_idx, edge_idx, alpha, segsum, Dd,
                                       efeat, acc, E);
  k_bnstats<<<512, 128, 0, stream>>>(acc, bias, colsum, colsumsq, N);
  k_bnparams<<<1, 128, 0, stream>>>(colsum, colsumsq, gamma, beta, scale,
                                    shift, N);
  long long total = (long long)N * FDIM;
  k_final<<<(unsigned)((total + 255) / 256), 256, 0, stream>>>(
      acc, bias, scale, shift, x, out, total);
}
